// Phonology1_25460566130713
// MI455X (gfx1250) — matrix operand order repaired and checked by reading, not compiled, not run on hardware
//
#include <hip/hip_runtime.h>
#include <cstddef>

typedef __attribute__((ext_vector_type(16))) _Float16 v16h;
typedef __attribute__((ext_vector_type(8)))  float    v8f;

#define RDIM 32   // roles R (== wave32 width)
#define FDIM 64   // F
#define PDIM 8    // P
#define CDIM 32   // C (context width == K of all GEMMs)

// ---------------- wave32 reductions ----------------
__device__ __forceinline__ float wave_sum(float v) {
#pragma unroll
  for (int off = 16; off >= 1; off >>= 1) v += __shfl_xor(v, off, 32);
  return v;
}
__device__ __forceinline__ float wave_max(float v) {
#pragma unroll
  for (int off = 16; off >= 1; off >>= 1) v = fmaxf(v, __shfl_xor(v, off, 32));
  return v;
}

// softmax weight for this lane's role at position `posn` (TAU = 1)
__device__ __forceinline__ float attn_w(float posn, float rr) {
  float d = posn - rr;
  float s = -d * d;
  float m = wave_max(s);
  float e = __expf(s - m);
  float sum = wave_sum(e);
  return e / sum;
}

// ---------------------------------------------------------------------------
// Y[M,N] = act(A[M,CDIM] @ W[CDIM,N] + bias[N]); one wave per 16x16 tile,
// K=32 consumed by a single v_wmma_f32_16x16x32_f16.
// ACT: 0 = identity, 1 = tanh
// ---------------------------------------------------------------------------
template <int ACT>
__global__ void gemm16_wmma(const float* __restrict__ A,
                            const float* __restrict__ W,
                            const float* __restrict__ bias,
                            float* __restrict__ Y, int M, int N) {
  const int lane = threadIdx.x & 31;
  const int wave = threadIdx.x >> 5;
  const int ntN  = N >> 4;
  const int tile = blockIdx.x * (blockDim.x >> 5) + wave;
  if (tile >= (M >> 4) * ntN) return;          // wave-uniform: EXEC stays full
  const int tm = tile / ntN;
  const int tn = tile - tm * ntN;
  const int hl = lane >> 4;                    // lane half (0/1)
  const int l  = lane & 15;

  // A fragment, 16x32 f16: lanes 0-15 row m=l hold K 0-15;
  // lanes 16-31 hold K 16-31.
  const float* arow = A + (size_t)(tm * 16 + l) * CDIM;
  const int kb = hl * 16;
  v16h af;
#pragma unroll
  for (int e = 0; e < 8; ++e) af[e] = (_Float16)arow[kb + e];
#pragma unroll
  for (int e = 0; e < 8; ++e) af[8 + e] = (_Float16)arow[kb + 8 + e];

  // B fragment, 32x16 f16: lane -> column (l); half element e -> K = hl*16+e.
  v16h bfr;
  const float* wcol = W + tn * 16 + l;
#pragma unroll
  for (int e = 0; e < 16; ++e) bfr[e] = (_Float16)wcol[(size_t)(hl * 16 + e) * N];

  v8f c = {};
  c = __builtin_amdgcn_wmma_f32_16x16x32_f16(false, af, false, bfr,
                                             (short)0, c, false, false);

  // D layout: VGPR v -> row (v + hl*8), col l.
  const float bv = bias[tn * 16 + l];
#pragma unroll
  for (int v = 0; v < 8; ++v) {
    float y = c[v] + bv;
    if (ACT == 1) y = tanhf(y);
    Y[(size_t)(tm * 16 + v + hl * 8) * N + tn * 16 + l] = y;
  }
}

// ---------------------------------------------------------------------------
// Per-batch prep: match = sigmoid(pattern^T form), change-softmax, w = exp,
// Z, apply_, and the four per-(b,r) mixing probabilities.
// One wave per batch element; lane = role r.
// ---------------------------------------------------------------------------
__global__ void prep_kernel(const float* __restrict__ form,      // B,F,R
                            const float* __restrict__ context,   // B,C
                            const float* __restrict__ Ww,        // C,P
                            const float* __restrict__ bw,        // P
                            const float* __restrict__ pattern,   // B,F*P (tanh)
                            const float* __restrict__ changeRaw, // B,4*P
                            float* __restrict__ applyW,          // B,R,P
                            float* __restrict__ pcoef) {         // B,R,4
  const int b = blockIdx.x;
  const int r = threadIdx.x;  // 0..31
  const float* fb = form    + (size_t)b * FDIM * RDIM;
  const float* pb = pattern + (size_t)b * FDIM * PDIM;

  // match[p] = sigmoid(sum_f pattern[f,p] * form[f,r])
  float match[PDIM];
#pragma unroll
  for (int p = 0; p < PDIM; ++p) match[p] = 0.f;
  for (int f = 0; f < FDIM; ++f) {
    const float fv = fb[f * RDIM + r];
#pragma unroll
    for (int p = 0; p < PDIM; ++p) match[p] = fmaf(pb[f * PDIM + p], fv, match[p]);
  }
#pragma unroll
  for (int p = 0; p < PDIM; ++p) match[p] = 1.f / (1.f + __expf(-match[p]));

  // w[p] = exp(context . Ww[:,p] + bw[p])   (wave-uniform)
  const float* cb = context + (size_t)b * CDIM;
  float wv[PDIM];
#pragma unroll
  for (int p = 0; p < PDIM; ++p) wv[p] = bw[p];
  for (int k = 0; k < CDIM; ++k) {
    const float cv = cb[k];
#pragma unroll
    for (int p = 0; p < PDIM; ++p) wv[p] = fmaf(cv, Ww[k * PDIM + p], wv[p]);
  }
#pragma unroll
  for (int p = 0; p < PDIM; ++p) wv[p] = __expf(wv[p]);

  // change = softmax over the 4 "change types" per p (wave-uniform)
  const float* crb = changeRaw + (size_t)b * 4 * PDIM;
  float ch[4][PDIM];
#pragma unroll
  for (int p = 0; p < PDIM; ++p) {
    float m = crb[p];
#pragma unroll
    for (int c2 = 1; c2 < 4; ++c2) m = fmaxf(m, crb[c2 * PDIM + p]);
    float s = 0.f;
#pragma unroll
    for (int c2 = 0; c2 < 4; ++c2) {
      ch[c2][p] = __expf(crb[c2 * PDIM + p] - m);
      s += ch[c2][p];
    }
    const float inv = 1.f / s;
#pragma unroll
    for (int c2 = 0; c2 < 4; ++c2) ch[c2][p] *= inv;
  }

  // Z, apply_, p_noch
  float Z = 1.f;
#pragma unroll
  for (int p = 0; p < PDIM; ++p) Z = fmaf(wv[p], match[p], Z);
  const float invZ = 1.f / Z;
  float ap[PDIM];
#pragma unroll
  for (int p = 0; p < PDIM; ++p) ap[p] = wv[p] * match[p] * invZ;

  float* aw = applyW + ((size_t)b * RDIM + r) * PDIM;
#pragma unroll
  for (int p = 0; p < PDIM; ++p) aw[p] = ap[p];

  float pm = 0.f, p1 = 0.f, p2 = 0.f;
#pragma unroll
  for (int p = 0; p < PDIM; ++p) {
    pm = fmaf(ap[p], ch[0][p], pm);  // change_all[...,0]
    p1 = fmaf(ap[p], ch[2][p], p1);  // change_all[...,2]
    p2 = fmaf(ap[p], ch[3][p], p2);  // change_all[...,3]
  }
  float* pc = pcoef + ((size_t)b * RDIM + r) * 4;
  pc[0] = pm; pc[1] = p1; pc[2] = p2; pc[3] = invZ;
}

// ---------------------------------------------------------------------------
// Sequential 32-step scan. One wave per batch; lane = role r; the (F=64,R=32)
// state lives in 64 VGPRs per lane. Posn is wave-uniform; the 5 positional
// softmaxes per step use wave32 shuffle reductions.
// ---------------------------------------------------------------------------
__global__ void scan_kernel(const float* __restrict__ form,    // B,F,R
                            const float* __restrict__ ftr1,    // B,F,P (tanh)
                            const float* __restrict__ ftr2,    // B,F,P (tanh)
                            const float* __restrict__ applyW,  // B,R,P
                            const float* __restrict__ pcoef,   // B,R,4
                            float* __restrict__ out) {         // B,F,R
  __shared__ float s_cp[FDIM], s_e1[FDIM], s_e2[FDIM];
  const int b    = blockIdx.x;
  const int lane = threadIdx.x;       // role r
  const float rr = (float)lane;

  const float* fb  = form   + (size_t)b * FDIM * RDIM;
  const float* f1b = ftr1   + (size_t)b * FDIM * PDIM;
  const float* f2b = ftr2   + (size_t)b * FDIM * PDIM;
  const float* awb = applyW + (size_t)b * RDIM * PDIM;
  const float* pcb = pcoef  + (size_t)b * RDIM * 4;

  float o[FDIM];
#pragma unroll
  for (int f = 0; f < FDIM; ++f) o[f] = 0.f;
  float posn = 0.f;

  for (int i = 0; i < RDIM; ++i) {
    // wave-uniform step scalars
    const float pm  = pcb[i * 4 + 0];
    const float p1  = pcb[i * 4 + 1];
    const float p2  = pcb[i * 4 + 2];
    const float pnc = pcb[i * 4 + 3];
    const float pcopy = pnc + p1 + p2;

    float apI[PDIM];
#pragma unroll
    for (int p = 0; p < PDIM; ++p) apI[p] = awb[i * PDIM + p];

    // build this step's filler rows cooperatively (2 f's per lane)
    __syncthreads();
#pragma unroll
    for (int j = 0; j < 2; ++j) {
      const int f = lane * 2 + j;
      float e1 = 0.f, e2 = 0.f;
#pragma unroll
      for (int p = 0; p < PDIM; ++p) {
        e1 = fmaf(apI[p], f1b[f * PDIM + p], e1);   // ftr1_all[b,i,f]
        e2 = fmaf(apI[p], f2b[f * PDIM + p], e2);   // ftr2_all[b,i,f]
      }
      s_e1[f] = e1;
      s_e2[f] = e2;
      s_cp[f] = fb[f * RDIM + i];                   // copy_all[b,i,f]
    }
    __syncthreads();

    // attention weights at the positions the reference visits
    const float a0 = attn_w(posn, rr);
    const float posn1 = posn + pcopy;
    const float a1 = attn_w(posn1, rr);
    const float posn2 = posn1 + p1;
    const float a2 = attn_w(posn2, rr);
    const float a3 = attn_w(posn2 + 1.f, rr);
    const float posn3 = posn2 + 2.f * p2;
    const float a4 = attn_w(posn3, rr);
    posn = posn3 + pm;

#pragma unroll
    for (int f = 0; f < FDIM; ++f) {
      const float cp = s_cp[f], e1 = s_e1[f], e2 = s_e2[f];
      const float md = fmaf(cp, 1.f - fabsf(e1), e1);  // mod_all
      float oo = o[f];
      float t  = fmaf(a0, cp - oo, oo);    // bind copy
      oo = fmaf(pcopy, t - oo, oo);
      t  = fmaf(a1, e1 - oo, oo);          // bind ep1 (first)
      oo = fmaf(p1, t - oo, oo);
      t  = fmaf(a2, e1 - oo, oo);          // bind ep1 (second)
      float u = fmaf(a3, e2 - t, t);       // then ep2
      oo = fmaf(p2, u - oo, oo);
      t  = fmaf(a4, md - oo, oo);          // bind mod
      oo = fmaf(pm, t - oo, oo);
      o[f] = oo;
    }
  }

#pragma unroll
  for (int f = 0; f < FDIM; ++f)
    out[(size_t)b * FDIM * RDIM + f * RDIM + lane] = o[f];
}

// ---------------------------------------------------------------------------
extern "C" void kernel_launch(void* const* d_in, const int* in_sizes, int n_in,
                              void* d_out, int out_size, void* d_ws, size_t ws_size,
                              hipStream_t stream) {
  (void)in_sizes; (void)n_in; (void)out_size; (void)ws_size;
  const float* form    = (const float*)d_in[0];
  const float* context = (const float*)d_in[1];
  const float* Wm = (const float*)d_in[2];
  const float* bm = (const float*)d_in[3];
  const float* Wc = (const float*)d_in[4];
  const float* bc = (const float*)d_in[5];
  const float* W1 = (const float*)d_in[6];
  const float* b1 = (const float*)d_in[7];
  const float* W2 = (const float*)d_in[8];
  const float* b2 = (const float*)d_in[9];
  const float* Ww = (const float*)d_in[10];
  const float* bw = (const float*)d_in[11];
  float* out = (float*)d_out;

  const int B = 4096;
  float* ws = (float*)d_ws;
  float* pattern   = ws;                              // B*512
  float* ftr1      = pattern   + (size_t)B * 512;     // B*512
  float* ftr2      = ftr1      + (size_t)B * 512;     // B*512
  float* changeRaw = ftr2      + (size_t)B * 512;     // B*32
  float* applyW    = changeRaw + (size_t)B * 32;      // B*256
  float* pcoef     = applyW    + (size_t)B * 256;     // B*128

  // context GEMMs via WMMA: M=4096 (256 tiles), K=32 (one wmma), N tiles vary
  const int WPB = 4;                  // 4 waves per block
  dim3 blk(32 * WPB);
  gemm16_wmma<1><<<(256 * 32) / WPB, blk, 0, stream>>>(context, Wm, bm, pattern, B, 512);
  gemm16_wmma<1><<<(256 * 32) / WPB, blk, 0, stream>>>(context, W1, b1, ftr1, B, 512);
  gemm16_wmma<1><<<(256 * 32) / WPB, blk, 0, stream>>>(context, W2, b2, ftr2, B, 512);
  gemm16_wmma<0><<<(256 * 2)  / WPB, blk, 0, stream>>>(context, Wc, bc, changeRaw, B, 32);

  prep_kernel<<<B, 32, 0, stream>>>(form, context, Ww, bw, pattern, changeRaw,
                                    applyW, pcoef);
  scan_kernel<<<B, 32, 0, stream>>>(form, ftr1, ftr2, applyW, pcoef, out);
}
